// RNNModel_40905268527589
// MI455X (gfx1250) — matrix operand order repaired and checked by reading, not compiled, not run on hardware
//
#include <hip/hip_runtime.h>
#include <hip/hip_bf16.h>

typedef __attribute__((ext_vector_type(16))) _Float16 v16h;
typedef __attribute__((ext_vector_type(8)))  _Float16 v8h;
typedef __attribute__((ext_vector_type(8)))  float    v8f;

#define BATCH 64
#define TSTEPS 256
#define HID 200
#define HPAD 224          // 200 padded to 7 k-tiles of 32
#define G4H 800           // 4*H
#define D1 350
#define D1P 352           // 11 k-tiles
#define D2 400
#define D2P 416           // 13 k-tiles
#define ROWS (BATCH*TSTEPS)   // 16384
#define NT_G 50           // 800/16 n-tiles

// LDS layout for the persistent LSTM kernel (284,672 B < 320 KB/WGP)
#define SM_Z_BYTES   ((size_t)BATCH * G4H * 4)     // 204800
#define SM_H_BYTES   ((size_t)BATCH * HPAD * 2)    // 28672
#define SM_C_BYTES   ((size_t)BATCH * HID * 4)     // 51200
#define SM_TOTAL     (SM_Z_BYTES + SM_H_BYTES + SM_C_BYTES)

__device__ __forceinline__ float sigm(float x) { return 1.0f / (1.0f + __expf(-x)); }

// ---------------------------------------------------------------------------
// Pack concat(tokens,tags) -> f16, padded to 352 cols
__global__ void pack_x_kernel(const float* __restrict__ tok,
                              const float* __restrict__ tag,
                              _Float16* __restrict__ X) {
    int id = blockIdx.x * blockDim.x + threadIdx.x;
    if (id >= ROWS * D1P) return;
    int row = id / D1P, c = id - row * D1P;
    float v = 0.0f;
    if (c < 300)      v = tok[(size_t)row * 300 + c];
    else if (c < 350) v = tag[(size_t)row * 50 + (c - 300)];
    X[id] = (_Float16)v;
}

// ---------------------------------------------------------------------------
// Pack a weight slice W[rowOff .. rowOff+rows) x cols (row stride ld) into
// WMMA B-fragment order: [kt][nt][lane][16]; element i of lane l is
// B[kt*32 + 8*(l/16) + (i&7) + 16*(i/8)][nt*16 + (l&15)]  — v_wmma 16-bit B layout.
__global__ void pack_bfrag_kernel(const float* __restrict__ W, int ld, int rowOff,
                                  int rows, int cols, int KT, int NT,
                                  _Float16* __restrict__ dst) {
    int id = blockIdx.x * blockDim.x + threadIdx.x;
    if (id >= KT * NT * 512) return;
    int i    = id & 15;
    int lane = (id >> 4) & 31;
    int nt   = (id >> 9) % NT;
    int kt   = id / (512 * NT);
    int k = kt * 32 + (lane >> 4) * 8 + (i & 7) + (i >> 3) * 16;
    int n = nt * 16 + (lane & 15);
    float v = (k < rows && n < cols) ? W[(size_t)(k + rowOff) * ld + n] : 0.0f;
    dst[id] = (_Float16)v;
}

// ---------------------------------------------------------------------------
// Generic big WMMA GEMM: C[M,Nreal] = A[M,KT*32](f16,row-major,lda) * Bp + bias
__global__ void gemm_wmma_kernel(const _Float16* __restrict__ A, int lda,
                                 const _Float16* __restrict__ Bp,
                                 const float* __restrict__ bias,
                                 float* __restrict__ C, int ldc,
                                 int M, int Nreal, int KT, int NT) {
    int lane = threadIdx.x & 31;
    int wave = blockIdx.x * (blockDim.x >> 5) + (threadIdx.x >> 5);
    int MT = M >> 4;
    if (wave >= MT * NT) return;
    int mt = wave / NT, nt = wave - mt * NT;
    int hf = lane >> 4, lm = lane & 15;
    const _Float16* Arow  = A + (size_t)(mt * 16 + lm) * lda + hf * 8;
    const _Float16* Bfrag = Bp + ((size_t)nt * 32 + lane) * 16;
    v8f acc = {0.f, 0.f, 0.f, 0.f, 0.f, 0.f, 0.f, 0.f};
    for (int kt = 0; kt < KT; ++kt) {
        v8h alo = *(const v8h*)(Arow + kt * 32);
        v8h ahi = *(const v8h*)(Arow + kt * 32 + 16);
        v16h a;
#pragma unroll
        for (int i = 0; i < 8; ++i) { a[i] = alo[i]; a[i + 8] = ahi[i]; }
        v16h b = *(const v16h*)(Bfrag + (size_t)kt * NT * 512);
        acc = __builtin_amdgcn_wmma_f32_16x16x32_f16(false, a, false, b,
                                                     (short)0, acc, false, false);
    }
    int col = nt * 16 + lm;
    if (col < Nreal) {
        float bv = bias ? bias[col] : 0.0f;
#pragma unroll
        for (int r = 0; r < 8; ++r) {
            int row = mt * 16 + r + hf * 8;
            C[(size_t)row * ldc + col] = acc[r] + bv;
        }
    }
}

// ---------------------------------------------------------------------------
// Persistent LSTM: one workgroup (1024 thr = 32 wave32) runs all 256 steps.
// State (h_prev f16, c f32, gates z f32) lives entirely in LDS (284.7 KB of
// the WGP's 320 KB). Per step: WMMA gate GEMM (A from LDS, packed B from L2,
// Zx[b, idx(s)] added after the k loop) -> barrier -> cell update -> barrier.
// blockIdx.x selects direction (0=fwd params, 1=bwd params) so layer-1
// fwd+bwd run concurrently on two WGPs from a single launch.
__global__ void lstm_persistent_kernel(const _Float16* __restrict__ WhP_f,
                                       const _Float16* __restrict__ WhP_b,
                                       const float* __restrict__ Zx_f,
                                       const float* __restrict__ Zx_b,
                                       const int* __restrict__ lens,
                                       _Float16* __restrict__ catOut, // may be null
                                       int catStride,
                                       float* __restrict__ f32Out) {  // may be null
    extern __shared__ char smem[];
    float*    zs  = (float*)smem;                                   // [64][800]
    _Float16* hps = (_Float16*)(smem + SM_Z_BYTES);                 // [64][224]
    float*    cs  = (float*)(smem + SM_Z_BYTES + SM_H_BYTES);       // [64][200]

    const int dir = blockIdx.x;
    const _Float16* WhP = dir ? WhP_b : WhP_f;
    const float*    Zx  = dir ? Zx_b  : Zx_f;
    const int catOff = dir * HID;

    const int tid  = threadIdx.x;
    const int lane = tid & 31;
    const int wv   = tid >> 5;
    const int hf   = lane >> 4, lm = lane & 15;

    // zero-init recurrent state (incl. h padding cols 200..223)
    for (int i = tid; i < BATCH * HPAD; i += 1024) hps[i] = (_Float16)0.0f;
    for (int i = tid; i < BATCH * HID;  i += 1024) cs[i] = 0.0f;
    __syncthreads();

    for (int s = 0; s < TSTEPS; ++s) {
        // ---- gate GEMM phase: 200 16x16 tiles over 32 waves ----
        for (int tile = wv; tile < 4 * NT_G; tile += 32) {
            int mt = tile & 3, nt = tile >> 2;
            const _Float16* Arow = hps + (size_t)(mt * 16 + lm) * HPAD + hf * 8;
            const _Float16* Bfrag = WhP + ((size_t)nt * 32 + lane) * 16;
            v8f acc = {0.f, 0.f, 0.f, 0.f, 0.f, 0.f, 0.f, 0.f};
#pragma unroll
            for (int kt = 0; kt < 7; ++kt) {
                v8h alo = *(const v8h*)(Arow + kt * 32);        // ds_load_b128
                v8h ahi = *(const v8h*)(Arow + kt * 32 + 16);
                v16h a;
#pragma unroll
                for (int i = 0; i < 8; ++i) { a[i] = alo[i]; a[i + 8] = ahi[i]; }
                v16h b = *(const v16h*)(Bfrag + (size_t)kt * NT_G * 512);
                acc = __builtin_amdgcn_wmma_f32_16x16x32_f16(false, a, false, b,
                                                             (short)0, acc,
                                                             false, false);
            }
            int col = nt * 16 + lm;
#pragma unroll
            for (int r = 0; r < 8; ++r) {
                int b = mt * 16 + r + hf * 8;
                int len = lens[b];
                int idx = dir ? (s < len ? len - 1 - s : s) : s;
                zs[b * G4H + col] =
                    acc[r] + Zx[((size_t)b * TSTEPS + idx) * G4H + col];
            }
        }
        __syncthreads();

        // ---- cell update phase: 64*200 elements over 1024 threads ----
        for (int id = tid; id < BATCH * HID; id += 1024) {
            int b = id / HID, c = id - b * HID;
            float zi = zs[b * G4H + c];
            float zj = zs[b * G4H + 200 + c];
            float zf = zs[b * G4H + 400 + c];
            float zo = zs[b * G4H + 600 + c];
            int len = lens[b];
            bool act = s < len;
            float cold = cs[id];
            float nc = cold * sigm(zf + 1.0f) + sigm(zi) * tanhf(zj);
            float nh = tanhf(nc) * sigm(zo);
            if (act) { cs[id] = nc; hps[b * HPAD + c] = (_Float16)nh; }
            float outv = act ? nh : 0.0f;
            int tout = (dir && act) ? (len - 1 - s) : s;
            size_t o = (size_t)b * TSTEPS + tout;
            if (catOut) catOut[o * catStride + catOff + c] = (_Float16)outv;
            if (f32Out) f32Out[o * HID + c] = outv;
        }
        __syncthreads();
    }
}

// ---------------------------------------------------------------------------
// Batch-norm statistics over (B,T) per feature (population variance)
__global__ void bn_stats_kernel(const float* __restrict__ h2,
                                float* __restrict__ mean, float* __restrict__ rstd) {
    __shared__ float ss[256], sq[256];
    int f = blockIdx.x, tid = threadIdx.x;
    float s = 0.f, q = 0.f;
    for (int i = tid; i < ROWS; i += 256) {
        float v = h2[(size_t)i * HID + f];
        s += v; q += v * v;
    }
    ss[tid] = s; sq[tid] = q;
    __syncthreads();
    for (int st = 128; st > 0; st >>= 1) {
        if (tid < st) { ss[tid] += ss[tid + st]; sq[tid] += sq[tid + st]; }
        __syncthreads();
    }
    if (tid == 0) {
        float m = ss[0] * (1.0f / ROWS);
        float v = sq[0] * (1.0f / ROWS) - m * m;
        mean[f] = m;
        rstd[f] = rsqrtf(v + 1e-3f);
    }
}

__global__ void bn_apply_kernel(const float* __restrict__ h2,
                                const float* __restrict__ mean,
                                const float* __restrict__ rstd,
                                const float* __restrict__ gamma,
                                const float* __restrict__ beta,
                                _Float16* __restrict__ h2n) {
    int id = blockIdx.x * blockDim.x + threadIdx.x;
    if (id >= ROWS * HPAD) return;
    int row = id / HPAD, c = id - row * HPAD;
    float v = 0.0f;
    if (c < HID)
        v = (h2[(size_t)row * HID + c] - mean[c]) * rstd[c] * gamma[c] + beta[c];
    h2n[id] = (_Float16)v;
}

// ---------------------------------------------------------------------------
extern "C" void kernel_launch(void* const* d_in, const int* in_sizes, int n_in,
                              void* d_out, int out_size, void* d_ws, size_t ws_size,
                              hipStream_t stream) {
    (void)in_sizes; (void)n_in; (void)out_size; (void)ws_size;
    const float* tokens = (const float*)d_in[0];
    const float* tags   = (const float*)d_in[1];
    const int*   lens   = (const int*)  d_in[2];
    const float* Wf     = (const float*)d_in[3];
    const float* bf     = (const float*)d_in[4];
    const float* Wb     = (const float*)d_in[5];
    const float* bb     = (const float*)d_in[6];
    const float* W2     = (const float*)d_in[7];
    const float* b2     = (const float*)d_in[8];
    const float* gamma  = (const float*)d_in[9];
    const float* beta   = (const float*)d_in[10];
    const float* Wd     = (const float*)d_in[11];
    const float* bd     = (const float*)d_in[12];
    float* out = (float*)d_out;

    // ---- carve workspace (256B aligned chunks) ----
    char* p = (char*)d_ws;
    auto carve = [&](size_t bytes) -> char* {
        char* r = p;
        p += (bytes + 255) & ~(size_t)255;
        return r;
    };
    _Float16* Xf16  = (_Float16*)carve((size_t)ROWS * D1P * 2);
    _Float16* WxfP  = (_Float16*)carve((size_t)11 * NT_G * 512 * 2);
    _Float16* WxbP  = (_Float16*)carve((size_t)11 * NT_G * 512 * 2);
    _Float16* WhfP  = (_Float16*)carve((size_t)7  * NT_G * 512 * 2);
    _Float16* WhbP  = (_Float16*)carve((size_t)7  * NT_G * 512 * 2);
    _Float16* Wx2P  = (_Float16*)carve((size_t)13 * NT_G * 512 * 2);
    _Float16* Wh2P  = (_Float16*)carve((size_t)7  * NT_G * 512 * 2);
    _Float16* WdP   = (_Float16*)carve((size_t)7  * 3    * 512 * 2);
    float*    Zxf   = (float*)carve((size_t)ROWS * G4H * 4);
    float*    Zxb   = (float*)carve((size_t)ROWS * G4H * 4);
    float*    Zx2   = (float*)carve((size_t)ROWS * G4H * 4);
    _Float16* Hcat  = (_Float16*)carve((size_t)ROWS * D2P * 2);
    float*    h2    = (float*)carve((size_t)ROWS * HID * 4);
    _Float16* h2n   = (_Float16*)carve((size_t)ROWS * HPAD * 2);
    float*    meanb = (float*)carve((size_t)HID * 4);
    float*    rstdb = (float*)carve((size_t)HID * 4);

    // Hcat padding cols (400..415) must be zero for the layer-2 GEMM
    hipMemsetAsync(Hcat, 0, (size_t)ROWS * D2P * 2, stream);

    // ---- pack inputs + weights ----
    {
        int total = ROWS * D1P;
        pack_x_kernel<<<(total + 255) / 256, 256, 0, stream>>>(tokens, tags, Xf16);
    }
    auto packB = [&](const float* W, int ld, int rowOff, int rows, int cols,
                     int KT, int NT, _Float16* dst) {
        int total = KT * NT * 512;
        pack_bfrag_kernel<<<(total + 255) / 256, 256, 0, stream>>>(W, ld, rowOff, rows,
                                                                   cols, KT, NT, dst);
    };
    packB(Wf, G4H, 0,   D1,  G4H, 11, NT_G, WxfP);
    packB(Wf, G4H, D1,  HID, G4H, 7,  NT_G, WhfP);
    packB(Wb, G4H, 0,   D1,  G4H, 11, NT_G, WxbP);
    packB(Wb, G4H, D1,  HID, G4H, 7,  NT_G, WhbP);
    packB(W2, G4H, 0,   D2,  G4H, 13, NT_G, Wx2P);
    packB(W2, G4H, D2,  HID, G4H, 7,  NT_G, Wh2P);
    packB(Wd, 40,  0,   HID, 40,  7,  3,    WdP);

    // ---- hoisted input projections (big parallel GEMMs, bulk of the FLOPs) ----
    {
        int waves = (ROWS / 16) * NT_G;          // 51200 tiles
        int blocks = waves / 8;                  // 8 wave32 per 256-thr block
        gemm_wmma_kernel<<<blocks, 256, 0, stream>>>(Xf16, D1P, WxfP, bf, Zxf, G4H,
                                                     ROWS, G4H, 11, NT_G);
        gemm_wmma_kernel<<<blocks, 256, 0, stream>>>(Xf16, D1P, WxbP, bb, Zxb, G4H,
                                                     ROWS, G4H, 11, NT_G);
    }

    // ---- layer 1: fwd + bwd persistent recurrences (2 WGPs, 1 launch) ----
    lstm_persistent_kernel<<<2, 1024, SM_TOTAL, stream>>>(
        WhfP, WhbP, Zxf, Zxb, lens, Hcat, D2P, nullptr);

    // ---- layer 2 input projection ----
    {
        int waves = (ROWS / 16) * NT_G;
        gemm_wmma_kernel<<<waves / 8, 256, 0, stream>>>(Hcat, D2P, Wx2P, b2, Zx2, G4H,
                                                        ROWS, G4H, 13, NT_G);
    }

    // ---- layer 2 persistent recurrence ----
    lstm_persistent_kernel<<<1, 1024, SM_TOTAL, stream>>>(
        Wh2P, Wh2P, Zx2, Zx2, lens, nullptr, 0, h2);

    // ---- batch norm + final dense (WMMA) ----
    bn_stats_kernel<<<HID, 256, 0, stream>>>(h2, meanb, rstdb);
    {
        int total = ROWS * HPAD;
        bn_apply_kernel<<<(total + 255) / 256, 256, 0, stream>>>(h2, meanb, rstdb,
                                                                 gamma, beta, h2n);
    }
    {
        int waves = (ROWS / 16) * 3;             // 3072 tiles
        gemm_wmma_kernel<<<waves / 8, 256, 0, stream>>>(h2n, HPAD, WdP, bd, out, 40,
                                                        ROWS, 40, 7, 3);
    }
}
